// QIM_22789096473379
// MI455X (gfx1250) — compile-verified, matrix operand order checked by reading, not run on hardware
//
#include <hip/hip_runtime.h>
#include <hip/hip_bf16.h>

// ---------------------------------------------------------------------------
// MI455X (gfx1250) transformer block: sigmoid-attention decoder layer.
// - All matmuls on v_wmma_f32_16x16x32_bf16 (wave32).
// - Flash-style streaming attention (sigmoid gate => pure accumulate).
// - K-tile and mask-tile staged via the Tensor Data Mover
//   (tensor_load_to_lds, TENSORcnt) -- gfx1250 async-tensor path.
// ---------------------------------------------------------------------------

#define BATCH 4
#define SEQ   2048
#define DMODEL 256
#define NHEAD 8
#define DHEAD 32
#define MTOT  (BATCH * SEQ)          // 8192

typedef __attribute__((ext_vector_type(16))) __bf16 v16bf;
typedef __attribute__((ext_vector_type(8)))  __bf16 v8bf;
typedef __attribute__((ext_vector_type(8)))  float  v8f;
typedef __attribute__((ext_vector_type(4)))  unsigned int u32x4;
typedef __attribute__((ext_vector_type(8)))  int i32x8;
typedef __attribute__((ext_vector_type(4)))  int i32x4;

static __device__ __forceinline__ v16bf cat8(v8bf lo, v8bf hi) {
    v16bf r;
#pragma unroll
    for (int i = 0; i < 8; ++i) { r[i] = lo[i]; r[i + 8] = hi[i]; }
    return r;
}

static __device__ __forceinline__ v8f wmma_bf16(v16bf a, v16bf b, v8f c) {
    return __builtin_amdgcn_wmma_f32_16x16x32_bf16(false, a, false, b,
                                                   (short)0, c, false, false);
}

// ---------------------------------------------------------------------------
// TDM: 2D tile load Global -> LDS.  D# packing per CDNA5 ISA ch.8 (§8.3/8.4):
//   group0: [1:0]=count=1, [63:32]=lds_addr, [120:64]=global_addr, [127:126]=2
//   group1: [17:16]=data_size, [79:48]=tensor_dim0, [111:80]=tensor_dim1,
//           [127:112]=tile_dim0, [143:128]=tile_dim1, [207:160]=dim0_stride
//   groups 2/3 zero (2D tensor).
// data_size code: 0=1B 1=2B 2=4B 3=8B. dims/strides in elements.
// ---------------------------------------------------------------------------
static __device__ __forceinline__ void tdm_load_2d(
    unsigned lds_off, const void* gptr, unsigned ds_code,
    unsigned tile_d0, unsigned tile_d1,
    unsigned tensor_d0, unsigned tensor_d1, unsigned long long stride0)
{
    unsigned long long ga = (unsigned long long)gptr;
    u32x4 g0;
    g0[0] = 1u;                                            // count=1, user D#
    g0[1] = lds_off;                                       // LDS byte address
    g0[2] = (unsigned)(ga & 0xFFFFFFFFu);                  // global_addr lo
    g0[3] = (unsigned)((ga >> 32) & 0x1FFFFFFu) | (2u << 30); // hi | type=2
    i32x8 g1;
    g1[0] = (int)((ds_code & 3u) << 16);                   // wg_mask=0, no pad
    g1[1] = (int)((tensor_d0 & 0xFFFFu) << 16);            // barrier_addr=0
    g1[2] = (int)(((tensor_d0 >> 16) & 0xFFFFu) | ((tensor_d1 & 0xFFFFu) << 16));
    g1[3] = (int)(((tensor_d1 >> 16) & 0xFFFFu) | ((tile_d0 & 0xFFFFu) << 16));
    g1[4] = (int)(tile_d1 & 0xFFFFu);                      // tile_dim2=0
    g1[5] = (int)(stride0 & 0xFFFFFFFFull);
    g1[6] = (int)((stride0 >> 32) & 0xFFFFull);            // dim1_stride=0
    g1[7] = 0;
    i32x4 z4 = {0, 0, 0, 0};
#if defined(__clang_major__) && (__clang_major__ >= 23)
    i32x8 z8 = {0, 0, 0, 0, 0, 0, 0, 0};
    __builtin_amdgcn_tensor_load_to_lds(g0, g1, z4, z4, z8, 0);
#else
    __builtin_amdgcn_tensor_load_to_lds(g0, g1, z4, z4, 0);
#endif
}

static __device__ __forceinline__ unsigned lds_offset_of(const void* p) {
    return (unsigned)(unsigned long long)(uintptr_t)p;     // low 32 = LDS off
}

// ---------------------------------------------------------------------------
// prep: x = boxes + qdt  (f32 + bf16 copies)
// ---------------------------------------------------------------------------
__global__ __launch_bounds__(256) void prep_x_kernel(
    const float* __restrict__ qdt, const float* __restrict__ boxes,
    float* __restrict__ xf, __bf16* __restrict__ xb)
{
    int i = blockIdx.x * 256 + threadIdx.x;
    float v = qdt[i] + boxes[i];
    xf[i] = v;
    xb[i] = (__bf16)v;
}

// ---------------------------------------------------------------------------
// prep: convert 6 weights f32[k][n] -> bf16 transposed Wt[n][k]
// ---------------------------------------------------------------------------
__global__ __launch_bounds__(256) void prep_w_kernel(
    const float* __restrict__ W0, const float* __restrict__ W1,
    const float* __restrict__ W2, const float* __restrict__ W3,
    const float* __restrict__ W4, const float* __restrict__ W5,
    __bf16* __restrict__ out)
{
    int idx = blockIdx.x * 256 + threadIdx.x;          // 6*65536 total
    int w = idx >> 16;
    int r = idx & 65535;
    int nn = r >> 8;
    int k  = r & 255;
    const float* Wp = (w == 0) ? W0 : (w == 1) ? W1 : (w == 2) ? W2
                    : (w == 3) ? W3 : (w == 4) ? W4 : W5;
    out[idx] = (__bf16)Wp[k * 256 + nn];
}

// ---------------------------------------------------------------------------
// Generic 256-K GEMM: C[M,256] = A[M,256](bf16) @ W[256,256] + bias
//   Block tile 128x64, 8 waves (2x4), wave tile 64x16, K-loop 8 x 32.
// Epilogues: 0 = bf16 permuted [B,H,N,DH]  1 = f32  2 = relu->bf16
// ---------------------------------------------------------------------------
template <int EP>
__global__ __launch_bounds__(256) void gemm256_kernel(
    const __bf16* __restrict__ A, const __bf16* __restrict__ Wt,
    const float* __restrict__ bias, float* __restrict__ outF,
    __bf16* __restrict__ outB)
{
    __shared__ alignas(16) __bf16 As[128 * 32];
    __shared__ alignas(16) __bf16 Ws[64 * 32];

    const int tid  = threadIdx.x;
    const int lane = tid & 31;
    const int w    = tid >> 5;
    const int wm   = w >> 2;
    const int wn   = w & 3;
    const int rowbase = blockIdx.x * 128;
    const int colbase = blockIdx.y * 64;
    const int kb8 = (lane >> 4) * 8;

    const v8f vzero = {0.f, 0.f, 0.f, 0.f, 0.f, 0.f, 0.f, 0.f};
    v8f acc[4];
#pragma unroll
    for (int f = 0; f < 4; ++f) acc[f] = vzero;

    for (int ks = 0; ks < 8; ++ks) {
        const int k0 = ks * 32;
        __syncthreads();
#pragma unroll
        for (int u = tid; u < 512; u += 256) {
            int r = u >> 2, c8 = (u & 3) * 8;
            *(uint4*)&As[r * 32 + c8] =
                *(const uint4*)&A[(rowbase + r) * 256 + k0 + c8];
        }
        {
            int r = tid >> 2, c8 = (tid & 3) * 8;
            *(uint4*)&Ws[r * 32 + c8] =
                *(const uint4*)&Wt[(colbase + r) * 256 + k0 + c8];
        }
        if (k0 + 32 < 256)   // global_prefetch_b8 toward next K tile
            __builtin_prefetch(&A[(rowbase + (tid & 127)) * 256 + k0 + 32], 0, 0);
        __syncthreads();

        const int nl = wn * 16 + (lane & 15);
        v16bf bf = cat8(*(const v8bf*)&Ws[nl * 32 + kb8],
                        *(const v8bf*)&Ws[nl * 32 + kb8 + 16]);
#pragma unroll
        for (int f = 0; f < 4; ++f) {
            int rl = wm * 64 + f * 16 + (lane & 15);
            v16bf af = cat8(*(const v8bf*)&As[rl * 32 + kb8],
                            *(const v8bf*)&As[rl * 32 + kb8 + 16]);
            acc[f] = wmma_bf16(af, bf, acc[f]);
        }
    }

    const int rup = (lane >> 4) << 3;
    const int col = colbase + wn * 16 + (lane & 15);
    const float bv = bias[col];
#pragma unroll
    for (int f = 0; f < 4; ++f) {
#pragma unroll
        for (int i = 0; i < 8; ++i) {
            int row = rowbase + wm * 64 + f * 16 + i + rup;
            float val = acc[f][i] + bv;
            if (EP == 1) {
                outF[row * 256 + col] = val;
            } else if (EP == 2) {
                outB[row * 256 + col] = (__bf16)fmaxf(val, 0.f);
            } else {
                int b = row >> 11, n = row & 2047;
                int h = col >> 5,  dh = col & 31;
                outB[((b * NHEAD + h) * SEQ + n) * DHEAD + dh] = (__bf16)val;
            }
        }
    }
}

// ---------------------------------------------------------------------------
// Streaming sigmoid attention.  q,k,v: bf16 [B,H,N,32].
//   Block = (qtile of 128 rows) x (b,h); 8 waves, 16 q rows each;
//   64 chunks of 32 kv.  K-tile + mask-tile via TDM (tensor_load_to_lds),
//   V staged transposed manually, P re-laid out via per-wave LDS bounce.
// ---------------------------------------------------------------------------
__global__ __launch_bounds__(256) void attn_kernel(
    const __bf16* __restrict__ qb, const __bf16* __restrict__ kb,
    const __bf16* __restrict__ vb, const int* __restrict__ mask,
    __bf16* __restrict__ Ob)
{
    __shared__ alignas(16) __bf16 Ks[32 * 32];      // [kv][dh]   (TDM)
    __shared__ alignas(16) __bf16 Vts[32 * 32];     // [dh][kv]   (manual T)
    __shared__ alignas(16) int    msk[128 * 32];    // [qrow][kv] (TDM)
    __shared__ alignas(16) __bf16 Ps[128 * 32];     // per-wave P tiles

    const int tid  = threadIdx.x;
    const int lane = tid & 31;
    const int w    = tid >> 5;
    const int qtile = blockIdx.x;
    const int bh    = blockIdx.y;               // b*H + h
    const int base  = bh * SEQ;
    const int kb8   = (lane >> 4) * 8;
    const int rup   = (lane >> 4) << 3;

    const unsigned ks_off  = lds_offset_of(&Ks[0]);
    const unsigned msk_off = lds_offset_of(&msk[0]);

    // Q fragment (16 rows x 32 dh) lives in registers for the whole kernel
    const int qrow = qtile * 128 + w * 16 + (lane & 15);
    v16bf qfrag = cat8(*(const v8bf*)&qb[(base + qrow) * 32 + kb8],
                       *(const v8bf*)&qb[(base + qrow) * 32 + kb8 + 16]);

    const v8f vzero = {0.f, 0.f, 0.f, 0.f, 0.f, 0.f, 0.f, 0.f};
    v8f o0 = vzero, o1 = vzero;

    for (int kc = 0; kc < 64; ++kc) {
        const int kv0 = kc * 32;
        __syncthreads();

        // --- TDM: K tile 32(kv) x 32(dh) bf16, and mask tile 128 x 32 int
        if (w == 0) {
            tdm_load_2d(ks_off, &kb[(size_t)(base + kv0) * 32],
                        /*2B*/1, /*tile*/32, 32, /*tensor*/32, 32, /*stride*/32);
            tdm_load_2d(msk_off, &mask[(size_t)(qtile * 128) * 2048 + kv0],
                        /*4B*/2, /*tile*/32, 128, /*tensor*/32, 128, /*stride*/2048);
        }

        // --- V chunk, stored transposed [dh][kv] (manual; TDM can't transpose)
        {
            int kv = tid >> 3, dh0 = (tid & 7) * 4;
            const __bf16* src = &vb[(base + kv0 + kv) * 32 + dh0];
#pragma unroll
            for (int j = 0; j < 4; ++j) Vts[(dh0 + j) * 32 + kv] = src[j];
        }

        if (w == 0) __builtin_amdgcn_s_wait_tensorcnt(0);
        __syncthreads();

        // scores: B-frag = K^T (lane col = kv, K-dim = dh contiguous)
        v16bf bk0, bk1;
        {
            int kvl0 = (lane & 15), kvl1 = 16 + (lane & 15);
            bk0 = cat8(*(const v8bf*)&Ks[kvl0 * 32 + kb8],
                       *(const v8bf*)&Ks[kvl0 * 32 + kb8 + 16]);
            bk1 = cat8(*(const v8bf*)&Ks[kvl1 * 32 + kb8],
                       *(const v8bf*)&Ks[kvl1 * 32 + kb8 + 16]);
        }
        v8f s0 = wmma_bf16(qfrag, bk0, vzero);
        v8f s1 = wmma_bf16(qfrag, bk1, vzero);

        // gate + relayout P through per-wave LDS region
        const int rb = w * 16;
#pragma unroll
        for (int t = 0; t < 2; ++t) {
            v8f s = t ? s1 : s0;
            int col = t * 16 + (lane & 15);
#pragma unroll
            for (int i = 0; i < 8; ++i) {
                int rr = i + rup;
                float sv = s[i] * 0.17677669529663687f;    // 1/sqrt(32)
                float p = msk[(rb + rr) * 32 + col]
                              ? 1.f / (1.f + __expf(-sv)) : 0.f;
                Ps[(rb + rr) * 32 + col] = (__bf16)p;
            }
        }
        // wave-internal cross-lane LDS RAW: drain DScnt before re-reading
        asm volatile("s_wait_dscnt 0" ::: "memory");

        v16bf pa = cat8(*(const v8bf*)&Ps[(rb + (lane & 15)) * 32 + kb8],
                        *(const v8bf*)&Ps[(rb + (lane & 15)) * 32 + kb8 + 16]);
        int dh0 = (lane & 15), dh1 = 16 + (lane & 15);
        v16bf bv0 = cat8(*(const v8bf*)&Vts[dh0 * 32 + kb8],
                         *(const v8bf*)&Vts[dh0 * 32 + kb8 + 16]);
        v16bf bv1 = cat8(*(const v8bf*)&Vts[dh1 * 32 + kb8],
                         *(const v8bf*)&Vts[dh1 * 32 + kb8 + 16]);
        o0 = wmma_bf16(pa, bv0, o0);
        o1 = wmma_bf16(pa, bv1, o1);
    }

    // write O as bf16 [B,N,D] row-major (A input for output projection)
    const int h = bh & 7, b = bh >> 3;
#pragma unroll
    for (int t = 0; t < 2; ++t) {
#pragma unroll
        for (int i = 0; i < 8; ++i) {
            int n = qtile * 128 + w * 16 + i + rup;
            int col = h * 32 + t * 16 + (lane & 15);
            Ob[(b * SEQ + n) * 256 + col] = (__bf16)(t ? o1[i] : o0[i]);
        }
    }
}

// ---------------------------------------------------------------------------
// LayerNorm(a + r) * g + be : one 256-thread block per row.
// ---------------------------------------------------------------------------
__global__ __launch_bounds__(256) void ln_kernel(
    const float* __restrict__ a, const float* __restrict__ r,
    const float* __restrict__ g, const float* __restrict__ be,
    float* __restrict__ outF, __bf16* __restrict__ outB)
{
    __shared__ float red[256];
    const int row = blockIdx.x, t = threadIdx.x;
    float v = a[row * 256 + t] + r[row * 256 + t];
    red[t] = v;
    __syncthreads();
    for (int s = 128; s > 0; s >>= 1) {
        if (t < s) red[t] += red[t + s];
        __syncthreads();
    }
    float mu = red[0] * (1.f / 256.f);
    __syncthreads();
    float d = v - mu;
    red[t] = d * d;
    __syncthreads();
    for (int s = 128; s > 0; s >>= 1) {
        if (t < s) red[t] += red[t + s];
        __syncthreads();
    }
    float var = red[0] * (1.f / 256.f);
    float o = d * rsqrtf(var + 1e-5f) * g[t] + be[t];
    outF[row * 256 + t] = o;
    if (outB) outB[row * 256 + t] = (__bf16)o;
}

// ---------------------------------------------------------------------------
extern "C" void kernel_launch(void* const* d_in, const int* in_sizes, int n_in,
                              void* d_out, int out_size, void* d_ws, size_t ws_size,
                              hipStream_t stream)
{
    (void)in_sizes; (void)n_in; (void)out_size; (void)ws_size;

    const float* qdt   = (const float*)d_in[0];
    const float* boxes = (const float*)d_in[1];
    const int*   mask  = (const int*)d_in[2];
    const float* Wq = (const float*)d_in[3];  const float* bq = (const float*)d_in[4];
    const float* Wk = (const float*)d_in[5];  const float* bk = (const float*)d_in[6];
    const float* Wv = (const float*)d_in[7];  const float* bv = (const float*)d_in[8];
    const float* Wo = (const float*)d_in[9];  const float* bo = (const float*)d_in[10];
    const float* W1 = (const float*)d_in[11]; const float* b1 = (const float*)d_in[12];
    const float* W2 = (const float*)d_in[13]; const float* b2 = (const float*)d_in[14];
    const float* g1 = (const float*)d_in[15]; const float* be1 = (const float*)d_in[16];
    const float* g2 = (const float*)d_in[17]; const float* be2 = (const float*)d_in[18];

    // workspace layout (aliased across stage lifetimes, ~30 MB)
    char* ws = (char*)d_ws;
    const size_t MB = 1024 * 1024;
    float*  xf32  = (float*)(ws + 0);                   // 8 MB (-> ff2 later)
    __bf16* xb16  = (__bf16*)(ws + 8 * MB);             // 4 MB (-> tgt1 bf16 later)
    __bf16* Wt    = (__bf16*)(ws + 12 * MB);            // 768 KB
    char*   p0    = ws + 12 * MB + 786432;              // 16 MB region:
    __bf16* qbuf  = (__bf16*)(p0);
    __bf16* kbuf  = (__bf16*)(p0 + 4 * MB);
    __bf16* vbuf  = (__bf16*)(p0 + 8 * MB);
    __bf16* Obuf  = (__bf16*)(p0 + 12 * MB);
    float*  attnout = (float*)(p0);                     // aliases q+k (8 MB)
    float*  tgt1f   = (float*)(p0 + 8 * MB);            // aliases v+O (8 MB)
    __bf16* tgt1b   = xb16;
    __bf16* ff1     = (__bf16*)(p0);
    float*  ff2     = xf32;
    float*  out     = (float*)d_out;

    const dim3 gGemm(MTOT / 128, DMODEL / 64);          // 64 x 4
    const dim3 gAttn(SEQ / 128, BATCH * NHEAD);         // 16 x 32

    prep_x_kernel<<<MTOT * DMODEL / 256, 256, 0, stream>>>(qdt, boxes, xf32, xb16);
    prep_w_kernel<<<6 * 65536 / 256, 256, 0, stream>>>(Wq, Wk, Wv, Wo, W1, W2, Wt);

    gemm256_kernel<0><<<gGemm, 256, 0, stream>>>(xb16, Wt + 0 * 65536, bq, nullptr, qbuf);
    gemm256_kernel<0><<<gGemm, 256, 0, stream>>>(xb16, Wt + 1 * 65536, bk, nullptr, kbuf);
    gemm256_kernel<0><<<gGemm, 256, 0, stream>>>(xb16, Wt + 2 * 65536, bv, nullptr, vbuf);

    attn_kernel<<<gAttn, 256, 0, stream>>>(qbuf, kbuf, vbuf, mask, Obuf);

    gemm256_kernel<1><<<gGemm, 256, 0, stream>>>(Obuf, Wt + 3 * 65536, bo, attnout, nullptr);

    ln_kernel<<<MTOT, 256, 0, stream>>>(xf32, attnout, g1, be1, tgt1f, tgt1b);

    gemm256_kernel<2><<<gGemm, 256, 0, stream>>>(tgt1b, Wt + 4 * 65536, b1, nullptr, ff1);
    gemm256_kernel<1><<<gGemm, 256, 0, stream>>>(ff1,   Wt + 5 * 65536, b2, ff2, nullptr);

    ln_kernel<<<MTOT, 256, 0, stream>>>(tgt1f, ff2, g2, be2, out, nullptr);
}
